// GatedGraphNeuralNetwork_8718783611244
// MI455X (gfx1250) — matrix operand order repaired and checked by reading, not compiled, not run on hardware
//
#include <hip/hip_runtime.h>
#include <hip/hip_bf16.h>
#include <math.h>

// ---------------------------------------------------------------------------
// GGNN for MI455X (gfx1250, wave32, WMMA + async global->LDS staging).
// h kept in f32 (d_out); GEMMs are bf16 WMMA 16x16x32 with f32 accumulation.
// ---------------------------------------------------------------------------

#define N_NODES  50000
#define HIDDEN   256
#define ANNOT    32
#define CONCATD  (HIDDEN + ANNOT)   // 288
#define N_ETYPES 4
#define N_EDGES  75000
#define N_LAYERS 2

#define LDS_PAD  8                  // bf16 elements of row padding (bank spread)
#define LDA256   (HIDDEN + LDS_PAD)   // 264 bf16 = 132 dwords; starts 4r mod 64
#define LDA288   (CONCATD + LDS_PAD)  // 296 bf16 = 148 dwords; starts 20r mod 64

typedef __bf16 bf16x16 __attribute__((ext_vector_type(16)));
typedef float  f32x8   __attribute__((ext_vector_type(8)));
typedef int    v4i     __attribute__((vector_size(16)));

union FragU { bf16x16 f; uint4 u[2]; };

// ---------------------------------------------------------------------------
// Async global->LDS copy of one 16-byte chunk (ASYNCcnt path on gfx1250),
// with a safe fallback if the builtin is unavailable on this toolchain.
// Builtin param types (from hipcc diagnostic): v4i* (global), v4i* (lds).
// ---------------------------------------------------------------------------
__device__ __forceinline__ void async_copy16(const __hip_bfloat16* g, __hip_bfloat16* l) {
#if __has_builtin(__builtin_amdgcn_global_load_async_to_lds_b128)
  typedef __attribute__((address_space(1))) v4i gv4i;
  typedef __attribute__((address_space(3))) v4i lv4i;
  __builtin_amdgcn_global_load_async_to_lds_b128((gv4i*)g, (lv4i*)l, 0, 0);
#else
  *reinterpret_cast<uint4*>(l) = *reinterpret_cast<const uint4*>(g);
#endif
}

__device__ __forceinline__ void async_join() {
#if __has_builtin(__builtin_amdgcn_global_load_async_to_lds_b128)
#if __has_builtin(__builtin_amdgcn_s_wait_asynccnt)
  __builtin_amdgcn_s_wait_asynccnt(0);
#endif
#endif
  __syncthreads();
}

// ---------------------------------------------------------------------------
// Fragment loaders (per ISA 7.12.2 layouts)
// ---------------------------------------------------------------------------
// A (16x32 MxK) from a padded LDS tile: lane m = lane&15, half = lane>>4;
// K = half*8+[0..7] then 16+half*8+[0..7]  -> two 16B ds loads.
__device__ __forceinline__ bf16x16 lds_frag_a(const __hip_bfloat16* tile, int ld,
                                              int k0, int lane) {
  const int r = lane & 15, hs = (lane >> 4) & 1;
  const __hip_bfloat16* p = tile + (size_t)r * ld + k0 + hs * 8;
  FragU f;
  f.u[0] = *reinterpret_cast<const uint4*>(p);
  f.u[1] = *reinterpret_cast<const uint4*>(p + 16);
  return f.f;
}

// B (32x16 KxN) from row-major W[N][K] in global: lane n = lane&15;
// K = half*8+[0..7] then 16+half*8+[0..7] (same order as A).
__device__ __forceinline__ bf16x16 load_frag_b(const __hip_bfloat16* wrow, int k0, int lane) {
  const int kb = ((lane >> 4) & 1) * 8;
  const __hip_bfloat16* p = wrow + k0 + kb;
  FragU f;
  f.u[0] = *reinterpret_cast<const uint4*>(p);
  f.u[1] = *reinterpret_cast<const uint4*>(p + 16);
  return f.f;
}

__device__ __forceinline__ f32x8 wmma_bf16(bf16x16 a, bf16x16 b, f32x8 c) {
  return __builtin_amdgcn_wmma_f32_16x16x32_bf16(false, a, false, b, (short)0, c,
                                                 false, false);
}

// ---------------------------------------------------------------------------
// Elementwise conversions
// ---------------------------------------------------------------------------
__global__ void __launch_bounds__(256)
k_f32_to_bf16(const float* __restrict__ in, __hip_bfloat16* __restrict__ out, int n) {
  int i = blockIdx.x * blockDim.x + threadIdx.x;
  if (i < n) out[i] = __float2bfloat16(in[i]);
}

__global__ void __launch_bounds__(256)
k_concat_bf16(const float* __restrict__ x, const float* __restrict__ ann,
              __hip_bfloat16* __restrict__ xa) {
  int i = blockIdx.x * blockDim.x + threadIdx.x;
  if (i >= N_NODES * CONCATD) return;
  int node = i / CONCATD;
  int c = i - node * CONCATD;
  float v = (c < HIDDEN) ? x[(size_t)node * HIDDEN + c]
                         : ann[(size_t)node * ANNOT + (c - HIDDEN)];
  xa[i] = __float2bfloat16(v);
}

// ---------------------------------------------------------------------------
// Init: h = [x|ann] @ W_hid^T + b_hid   (M=50000, K=288, N=256)
// Block = 512 thr (16 waves) = one 16-node row; A tile staged once in LDS
// and shared by all 16 waves (wave w computes N-tile w).
// ---------------------------------------------------------------------------
__global__ void __launch_bounds__(512)
k_init_gemm(const __hip_bfloat16* __restrict__ xa, const __hip_bfloat16* __restrict__ W,
            const float* __restrict__ bias, float* __restrict__ h) {
  __shared__ __align__(16) __hip_bfloat16 sX[16 * LDA288];
  const int node0 = blockIdx.x * 16;

  // stage 16 x 288 bf16 (36 16B-chunks per row)
  for (int c = threadIdx.x; c < 16 * 36; c += 512) {
    int row = c / 36, cc = c - row * 36;
    async_copy16(xa + (size_t)(node0 + row) * CONCATD + cc * 8,
                 &sX[row * LDA288 + cc * 8]);
  }
  async_join();

  const int lane = threadIdx.x & 31;
  const int tn = threadIdx.x >> 5;            // 0..15
  const int r = lane & 15;
  const __hip_bfloat16* wrow = W + (size_t)(tn * 16 + r) * CONCATD;
  float b = bias[tn * 16 + r];
  f32x8 acc = {b, b, b, b, b, b, b, b};
#pragma unroll
  for (int k0 = 0; k0 < CONCATD; k0 += 32)
    acc = wmma_bf16(lds_frag_a(sX, LDA288, k0, lane), load_frag_b(wrow, k0, lane), acc);

  const int mb = ((lane >> 4) & 1) * 8;
  float* out = h + (size_t)(node0 + mb) * HIDDEN + tn * 16 + r;
#pragma unroll
  for (int v = 0; v < 8; ++v) out[(size_t)v * HIDDEN] = acc[v];
}

// ---------------------------------------------------------------------------
// Messages: msgs = h[src] @ W_msg[t]^T + b_msg[t], scatter-added with hw f32
// atomics. Block = 128 thr (4 waves) = 16 edges x 256 cols; the gathered
// A tile is async-staged to LDS once, each wave holds a 64-col strip
// (4 accumulators), so the gather traffic is paid once per 256 columns.
// ---------------------------------------------------------------------------
__global__ void __launch_bounds__(128)
k_msg_scatter(const __hip_bfloat16* __restrict__ hb,
              const __hip_bfloat16* __restrict__ Wm,   // [4][256][256] layer slice
              const float* __restrict__ bm,            // [4][256]
              const int* __restrict__ edges,           // [4][75000][2]
              float* __restrict__ incoming) {
  __shared__ __align__(16) __hip_bfloat16 sA[16 * LDA256];
  const int TME = (N_EDGES + 15) / 16;                 // 4688
  const int t = blockIdx.x / TME;
  const int tm = blockIdx.x - t * TME;

  // stage gathered rows: 16 rows x 32 chunks of 16B
  for (int c = threadIdx.x; c < 16 * 32; c += 128) {
    int row = c >> 5, cc = c & 31;
    int e = tm * 16 + row;
    if (e >= N_EDGES) e = N_EDGES - 1;                 // clamped tail gather
    int src = edges[((size_t)t * N_EDGES + e) * 2 + 0];
    async_copy16(hb + (size_t)src * HIDDEN + cc * 8, &sA[row * LDA256 + cc * 8]);
  }
  async_join();

  const int lane = threadIdx.x & 31;
  const int wv = threadIdx.x >> 5;                     // 0..3 -> 64-col strip
  const int r = lane & 15;

  const __hip_bfloat16* wrow[4];
  f32x8 acc[4];
#pragma unroll
  for (int sub = 0; sub < 4; ++sub) {
    int n0 = wv * 64 + sub * 16;
    float b = bm[t * HIDDEN + n0 + r];
    f32x8 z = {b, b, b, b, b, b, b, b};
    acc[sub] = z;
    wrow[sub] = Wm + ((size_t)t * HIDDEN + n0 + r) * HIDDEN;
  }
#pragma unroll
  for (int k0 = 0; k0 < HIDDEN; k0 += 32) {
    bf16x16 a = lds_frag_a(sA, LDA256, k0, lane);
#pragma unroll
    for (int sub = 0; sub < 4; ++sub)
      acc[sub] = wmma_bf16(a, load_frag_b(wrow[sub], k0, lane), acc[sub]);
  }

  const int mb = ((lane >> 4) & 1) * 8;
#pragma unroll
  for (int sub = 0; sub < 4; ++sub) {
    const int n = wv * 64 + sub * 16 + r;
#pragma unroll
    for (int v = 0; v < 8; ++v) {
      int em = tm * 16 + mb + v;
      if (em < N_EDGES) {
        int tgt = edges[((size_t)t * N_EDGES + em) * 2 + 1];
        unsafeAtomicAdd(&incoming[(size_t)tgt * HIDDEN + n], acc[sub][v]);
      }
    }
  }
}

// ---------------------------------------------------------------------------
// Fused GRU. Block = 512 thr (16 waves) = one 16-node row x 256 cols.
// inc/h A-tiles async-staged once to LDS, shared by all 16 waves; each wave
// owns one 16x16 output tile with 6 WMMA accumulators (r/z/n x ih/hh),
// gates evaluated in registers, h updated in place.
// ---------------------------------------------------------------------------
__global__ void __launch_bounds__(512)
k_gru(const __hip_bfloat16* __restrict__ incb, const __hip_bfloat16* __restrict__ hb,
      const __hip_bfloat16* __restrict__ Wih,  // [768][256] layer slice
      const __hip_bfloat16* __restrict__ Whh,  // [768][256] layer slice
      const float* __restrict__ bih, const float* __restrict__ bhh,
      float* __restrict__ h) {
  __shared__ __align__(16) __hip_bfloat16 sI[16 * LDA256];
  __shared__ __align__(16) __hip_bfloat16 sH[16 * LDA256];
  const int node0 = blockIdx.x * 16;

  {
    int c = threadIdx.x;                                // 512 chunks per tile
    int row = c >> 5, cc = c & 31;
    size_t goff = (size_t)(node0 + row) * HIDDEN + cc * 8;
    int loff = row * LDA256 + cc * 8;
    async_copy16(incb + goff, &sI[loff]);
    async_copy16(hb + goff, &sH[loff]);
  }
  async_join();

  const int lane = threadIdx.x & 31;
  const int tn = threadIdx.x >> 5;                      // 0..15
  const int r = lane & 15;
  const int col = tn * 16 + r;

  const __hip_bfloat16* wir = Wih + (size_t)(0 * HIDDEN + col) * HIDDEN;
  const __hip_bfloat16* wiz = Wih + (size_t)(1 * HIDDEN + col) * HIDDEN;
  const __hip_bfloat16* win = Wih + (size_t)(2 * HIDDEN + col) * HIDDEN;
  const __hip_bfloat16* whr = Whh + (size_t)(0 * HIDDEN + col) * HIDDEN;
  const __hip_bfloat16* whz = Whh + (size_t)(1 * HIDDEN + col) * HIDDEN;
  const __hip_bfloat16* whn = Whh + (size_t)(2 * HIDDEN + col) * HIDDEN;

  f32x8 air = {0,0,0,0,0,0,0,0}, aiz = air, ain = air;
  f32x8 ahr = air, ahz = air, ahn = air;
#pragma unroll
  for (int k0 = 0; k0 < HIDDEN; k0 += 32) {
    bf16x16 ai = lds_frag_a(sI, LDA256, k0, lane);
    bf16x16 ah = lds_frag_a(sH, LDA256, k0, lane);
    air = wmma_bf16(ai, load_frag_b(wir, k0, lane), air);
    aiz = wmma_bf16(ai, load_frag_b(wiz, k0, lane), aiz);
    ain = wmma_bf16(ai, load_frag_b(win, k0, lane), ain);
    ahr = wmma_bf16(ah, load_frag_b(whr, k0, lane), ahr);
    ahz = wmma_bf16(ah, load_frag_b(whz, k0, lane), ahz);
    ahn = wmma_bf16(ah, load_frag_b(whn, k0, lane), ahn);
  }

  const float bir = bih[col], biz = bih[HIDDEN + col], bin_ = bih[2 * HIDDEN + col];
  const float bhr = bhh[col], bhz = bhh[HIDDEN + col], bhn = bhh[2 * HIDDEN + col];
  const int mb = ((lane >> 4) & 1) * 8;
#pragma unroll
  for (int v = 0; v < 8; ++v) {
    size_t idx = (size_t)(node0 + mb + v) * HIDDEN + col;
    float gr = air[v] + bir + ahr[v] + bhr;
    float gz = aiz[v] + biz + ahz[v] + bhz;
    float rg = 1.0f / (1.0f + __expf(-gr));
    float zg = 1.0f / (1.0f + __expf(-gz));
    float ng = tanhf(ain[v] + bin_ + rg * (ahn[v] + bhn));
    float ho = h[idx];
    h[idx] = (1.0f - zg) * ng + zg * ho;
  }
}

// ---------------------------------------------------------------------------
// Host orchestration
// ---------------------------------------------------------------------------
extern "C" void kernel_launch(void* const* d_in, const int* in_sizes, int n_in,
                              void* d_out, int out_size, void* d_ws, size_t ws_size,
                              hipStream_t stream) {
  const float* x     = (const float*)d_in[0];
  const float* ann   = (const float*)d_in[1];
  const int*   edges = (const int*)d_in[2];
  const float* Whid  = (const float*)d_in[3];
  const float* bhid  = (const float*)d_in[4];
  const float* Wmsg  = (const float*)d_in[5];
  const float* bmsg  = (const float*)d_in[6];
  const float* Wih   = (const float*)d_in[7];
  const float* Whh   = (const float*)d_in[8];
  const float* bih   = (const float*)d_in[9];
  const float* bhh   = (const float*)d_in[10];
  float* h = (float*)d_out;

  char* ws = (char*)d_ws;
  size_t off = 0;
  auto take = [&](size_t bytes) -> void* {
    void* p = (void*)(ws + off);
    off += (bytes + 255) & ~(size_t)255;
    return p;
  };
  float*          incoming = (float*)take((size_t)N_NODES * HIDDEN * sizeof(float));
  __hip_bfloat16* hb       = (__hip_bfloat16*)take((size_t)N_NODES * HIDDEN * 2);
  __hip_bfloat16* incb     = (__hip_bfloat16*)take((size_t)N_NODES * HIDDEN * 2);
  __hip_bfloat16* xab      = (__hip_bfloat16*)take((size_t)N_NODES * CONCATD * 2);
  __hip_bfloat16* Whid_b   = (__hip_bfloat16*)take((size_t)HIDDEN * CONCATD * 2);
  __hip_bfloat16* Wmsg_b   = (__hip_bfloat16*)take((size_t)N_LAYERS * N_ETYPES * HIDDEN * HIDDEN * 2);
  __hip_bfloat16* Wih_b    = (__hip_bfloat16*)take((size_t)N_LAYERS * 3 * HIDDEN * HIDDEN * 2);
  __hip_bfloat16* Whh_b    = (__hip_bfloat16*)take((size_t)N_LAYERS * 3 * HIDDEN * HIDDEN * 2);

  auto cvt = [&](const float* in, __hip_bfloat16* out, int n) {
    k_f32_to_bf16<<<(n + 255) / 256, 256, 0, stream>>>(in, out, n);
  };
  cvt(Whid, Whid_b, HIDDEN * CONCATD);
  cvt(Wmsg, Wmsg_b, N_LAYERS * N_ETYPES * HIDDEN * HIDDEN);
  cvt(Wih,  Wih_b,  N_LAYERS * 3 * HIDDEN * HIDDEN);
  cvt(Whh,  Whh_b,  N_LAYERS * 3 * HIDDEN * HIDDEN);

  k_concat_bf16<<<(N_NODES * CONCATD + 255) / 256, 256, 0, stream>>>(x, ann, xab);
  k_init_gemm<<<N_NODES / 16, 512, 0, stream>>>(xab, Whid_b, bhid, h);

  const int TME = (N_EDGES + 15) / 16;
  const int steps[N_LAYERS] = {3, 3};

  for (int layer = 0; layer < N_LAYERS; ++layer) {
    for (int st = 0; st < steps[layer]; ++st) {
      cvt(h, hb, N_NODES * HIDDEN);
      (void)hipMemsetAsync(incoming, 0, (size_t)N_NODES * HIDDEN * sizeof(float), stream);
      k_msg_scatter<<<N_ETYPES * TME, 128, 0, stream>>>(
          hb,
          Wmsg_b + (size_t)layer * N_ETYPES * HIDDEN * HIDDEN,
          bmsg + (size_t)layer * N_ETYPES * HIDDEN,
          edges, incoming);
      cvt(incoming, incb, N_NODES * HIDDEN);
      k_gru<<<N_NODES / 16, 512, 0, stream>>>(
          incb, hb,
          Wih_b + (size_t)layer * 3 * HIDDEN * HIDDEN,
          Whh_b + (size_t)layer * 3 * HIDDEN * HIDDEN,
          bih + (size_t)layer * 3 * HIDDEN,
          bhh + (size_t)layer * 3 * HIDDEN,
          h);
    }
  }
}
